// HybridQLSTM_65481071405860
// MI455X (gfx1250) — matrix operand order repaired and checked by reading, not compiled, not run on hardware
//
#include <hip/hip_runtime.h>
#include <hip/hip_bf16.h>

// ---------------- problem sizes ----------------
#define T_STEPS 1024
#define BATCH   128
#define IN_DIM  256
#define HID     512
#define KDIM    768           // IN_DIM + HID
#define NWG     16            // persistent workgroups (== one cluster's worth)
#define KT      24            // KDIM/32 K-tiles
#define LDX     264           // padded x row stride (bf16) : 256 + 8  (TDM pad 4 DW)
#define LDH     520           // padded h row stride (bf16) : 512 + 8  (TDM pad 4 DW)

typedef __bf16 v16bf __attribute__((ext_vector_type(16)));
typedef __bf16 v8bf  __attribute__((ext_vector_type(8)));
typedef float  v8f   __attribute__((ext_vector_type(8)));
typedef unsigned int u32x4 __attribute__((ext_vector_type(4)));
typedef int          i32x8 __attribute__((ext_vector_type(8)));
typedef int          i32x4 __attribute__((ext_vector_type(4)));

// ---------------- LDS layout (bytes) ----------------
#define LDS_X_OFF   0u
#define LDS_X_BYTES (BATCH * LDX * 2u)                 // 67584
#define LDS_H_OFF   (LDS_X_OFF + LDS_X_BYTES)
#define LDS_H_BYTES (BATCH * LDH * 2u)                 // 133120
#define LDS_WF_OFF  (LDS_H_OFF + LDS_H_BYTES)          // 200704
#define LDS_TOTAL   (LDS_WF_OFF + (KDIM * 2 + 2 * BATCH + BATCH) * 4u)

// ---------------- workspace layout (bytes) ----------------
#define W_OFF   0ull
#define X_OFF   (4ull << 20)
#define X_BYTES ((unsigned long long)T_STEPS * BATCH * IN_DIM * 2ull)
#define H_OFF   (X_OFF + X_BYTES)
#define H_BYTES (2ull * BATCH * HID * 2ull)
#define BAR_OFF (H_OFF + H_BYTES)

// =====================================================================
// Kernel 1: fully parallel prep (HBM-streaming)
//  - X fp32 -> bf16
//  - Wi|Wu|Wo fp32 -> bf16, repacked into per-lane WMMA-B fragment order:
//    frag f = ntile*24+ktile holds a 32(K)x16(N) tile; lane L's 16 elems
//    are contiguous (32B): VGPR v half h  <->  K = 2v+h + (L<16?0:16), N = L&15
//  - zero h double-buffer and barrier counters
// =====================================================================
__global__ __launch_bounds__(256) void qlstm_prep(
    const float* __restrict__ X, const float* __restrict__ Wi,
    const float* __restrict__ Wu, const float* __restrict__ Wo,
    unsigned short* __restrict__ WfragU, unsigned short* __restrict__ XbU,
    unsigned short* __restrict__ hbU, unsigned int* __restrict__ bar) {
  __bf16* Wfrag = (__bf16*)WfragU;
  __bf16* Xb = (__bf16*)XbU;

  const long XN = (long)T_STEPS * BATCH * IN_DIM;  // 33554432
  const long WN = (long)KDIM * 1536;               // 1179648
  const long HN = 2L * BATCH * HID;                // 131072
  const long total = XN + WN + HN + 64;

  for (long idx = (long)blockIdx.x * 256 + threadIdx.x; idx < total;
       idx += (long)gridDim.x * 256) {
    if (idx < XN) {
      Xb[idx] = (__bf16)X[idx];
    } else if (idx < XN + WN) {
      long e = idx - XN;
      int f = (int)(e >> 9);
      int r = (int)(e & 511);
      int lane = r >> 4;
      int q = r & 15;
      int kloc = q + ((lane < 16) ? 0 : 16);
      int nloc = lane & 15;
      int ntile = f / KT;
      int ktile = f - ntile * KT;
      int kk = ktile * 32 + kloc;      // 0..767
      int nn = ntile * 16 + nloc;      // 0..1535 over [i|u|o]
      int mat = nn >> 9;
      int col = nn & 511;
      const float* Ws = (mat == 0) ? Wi : ((mat == 1) ? Wu : Wo);
      Wfrag[e] = (__bf16)Ws[(long)kk * HID + col];
    } else if (idx < XN + WN + HN) {
      hbU[idx - (XN + WN)] = 0;  // bf16 +0.0
    } else {
      bar[idx - (XN + WN + HN)] = 0u;
    }
  }
}

// =====================================================================
// TDM: 2D tile DMA global->LDS with hardware row padding (pad 4 DWORDs).
// D# packing per CDNA5 ISA 8.3/8.4. Issued once per workgroup by wave 0,
// tracked by TENSORcnt. clang-23 6-arg builtin; trailing groups zero (<=2D).
// =====================================================================
__device__ __forceinline__ void tdm_load_2d(unsigned lds_off, const void* gptr,
                                            unsigned dim0, unsigned rows,
                                            unsigned pad_interval_code) {
  unsigned long long ga = (unsigned long long)(uintptr_t)gptr;
  u32x4 g0;
  g0[0] = 1u;                                        // count=1, user descriptor
  g0[1] = lds_off;                                   // lds_addr
  g0[2] = (unsigned)ga;                              // global_addr[31:0]
  g0[3] = (unsigned)(ga >> 32) | (2u << 30);         // global_addr[56:32] | type=2
  i32x8 g1;
  g1[0] = (int)((1u << 16) |                         // data_size = 2 bytes
                (1u << 20) |                         // pad_enable
                (pad_interval_code << 22) |          // pad interval code
                (3u << 25));                         // pad_amount = 4 DWORDs
  g1[1] = (int)(dim0 << 16);                         // tensor_dim0[15:0]
  g1[2] = (int)(rows << 16);                         // dim0[31:16]=0 | tensor_dim1[15:0]
  g1[3] = (int)(dim0 << 16);                         // dim1[31:16]=0 | tile_dim0
  g1[4] = (int)rows;                                 // tile_dim1 | tile_dim2=0
  g1[5] = (int)dim0;                                 // tensor_dim0_stride[31:0]
  g1[6] = 0;                                         // stride0[47:32] | stride1[15:0]
  g1[7] = 0;                                         // stride1[47:16]
  i32x4 z4;
  z4[0] = 0; z4[1] = 0; z4[2] = 0; z4[3] = 0;
  i32x8 z8;
  z8[0] = 0; z8[1] = 0; z8[2] = 0; z8[3] = 0;
  z8[4] = 0; z8[5] = 0; z8[6] = 0; z8[7] = 0;
  __builtin_amdgcn_tensor_load_to_lds(g0, g1, z4, z4, z8, 0);
}

// =====================================================================
// Kernel 2: persistent recurrent phase. 16 WGs x 256 threads (8 wave32).
// =====================================================================
__device__ __forceinline__ void grid_sync(unsigned int* cnt, unsigned int* gen,
                                          unsigned int nwg) {
  __threadfence();   // release this WG's h stores to agent scope
  __syncthreads();
  if (threadIdx.x == 0) {
    unsigned int g = __hip_atomic_load(gen, __ATOMIC_RELAXED, __HIP_MEMORY_SCOPE_AGENT);
    unsigned int a = __hip_atomic_fetch_add(cnt, 1u, __ATOMIC_ACQ_REL, __HIP_MEMORY_SCOPE_AGENT);
    if (a == nwg - 1u) {
      __hip_atomic_store(cnt, 0u, __ATOMIC_RELAXED, __HIP_MEMORY_SCOPE_AGENT);
      __hip_atomic_store(gen, g + 1u, __ATOMIC_RELEASE, __HIP_MEMORY_SCOPE_AGENT);
    } else {
      while (__hip_atomic_load(gen, __ATOMIC_ACQUIRE, __HIP_MEMORY_SCOPE_AGENT) == g) {
        __builtin_amdgcn_s_sleep(2);
      }
    }
  }
  __syncthreads();
  // CDNA5 cluster barrier: real HW sync when launched as a 16-WG cluster,
  // architected NOP (ClusterID==0) otherwise.
  __builtin_amdgcn_s_cluster_barrier();
}

__global__ __launch_bounds__(256, 1) void qlstm_rnn(
    const unsigned short* __restrict__ WfragU, const unsigned short* __restrict__ XbU,
    unsigned short* __restrict__ hbU, unsigned int* __restrict__ bar,
    const float* __restrict__ Wf, const float* __restrict__ bfv,
    const float* __restrict__ bi, const float* __restrict__ bu,
    const float* __restrict__ bo, const float* __restrict__ phiv,
    float* __restrict__ out) {
  extern __shared__ char smem[];
  __bf16* xLDS = (__bf16*)(smem + LDS_X_OFF);     // 128 x 264 bf16 (TDM-padded)
  __bf16* hLDS = (__bf16*)(smem + LDS_H_OFF);     // 128 x 520 bf16 (TDM-padded)
  float* wfL = (float*)(smem + LDS_WF_OFF);       // 768*2 f32
  float* aL = wfL + KDIM * 2;                     // 128*2 f32
  float* fL = aL + 2 * BATCH;                     // 128 f32

  const __bf16* Wfrag = (const __bf16*)WfragU;
  const __bf16* Xb = (const __bf16*)XbU;
  __bf16* hb = (__bf16*)hbU;

  const int tid = threadIdx.x;
  const int g = blockIdx.x;
  const int lane = tid & 31;
  const int wave = tid >> 5;                 // M-tile (8 waves x 16 rows = 128)
  const int nlo = lane & 15;                 // N within tile / C-layout column
  const int khalf = (lane < 16) ? 0 : 16;    // A-operand K sub-chunk base
  const int mrow_off = (lane < 16) ? 0 : 8;  // C-layout row offset

  // Wf (768x2) into LDS once
  for (int e = tid; e < KDIM * 2; e += 256) wfL[e] = Wf[e];

  // per-thread constants for f-gate (thread -> (row, column j))
  const int frow = tid & 127, fj = tid >> 7;
  const float bfj = bfv[fj], phij = phiv[fj];

  // per-lane biases, one scalar per N-tile (C layout: all 8 VGPRs share N)
  float biasReg[6];
  int gnt[6];  // global ntile index into Wfrag per N-tile
  {
#pragma unroll
    for (int mat = 0; mat < 3; ++mat) {
      const float* bp = (mat == 0) ? bi : ((mat == 1) ? bu : bo);
#pragma unroll
      for (int sub = 0; sub < 2; ++sub) {
        biasReg[mat * 2 + sub] = bp[g * 32 + sub * 16 + nlo];
        gnt[mat * 2 + sub] = mat * 32 + g * 2 + sub;
      }
    }
  }

  // cell state lives in accumulator-layout registers for the whole scan
  float cst[2][8];
#pragma unroll
  for (int s = 0; s < 2; ++s)
#pragma unroll
    for (int r = 0; r < 8; ++r) cst[s][r] = 0.f;

  __syncthreads();  // wfL ready before first f-gate

  for (int t = 0; t < T_STEPS; ++t) {
    if (t + 1 < T_STEPS)  // global_prefetch_b8 of next step's x tile
      __builtin_prefetch((const void*)(Xb + (size_t)(t + 1) * BATCH * IN_DIM), 0, 1);

    // ---- TDM: DMA x_t (128x256) and h_t (128x512) tiles into padded LDS ----
    if (tid < 32) {
      const __bf16* xs = Xb + (size_t)t * BATCH * IN_DIM;
      const __bf16* hs = hb + (size_t)(t & 1) * BATCH * HID;
      tdm_load_2d(LDS_X_OFF, (const void*)xs, IN_DIM, BATCH, 6u);  // 128-DW rows
      tdm_load_2d(LDS_H_OFF, (const void*)hs, HID, BATCH, 7u);     // 256-DW rows
      __builtin_amdgcn_s_wait_tensorcnt(0);
    }
    __syncthreads();

    // ---- f-gate: softmax over 2 logits collapses to sigmoid(a0-a1) ----
    {
      float accf = 0.f;
      for (int k = 0; k < IN_DIM; k += 8) {
        v8bf cv = *(const v8bf*)&xLDS[frow * LDX + k];
#pragma unroll
        for (int u = 0; u < 8; ++u) accf += (float)cv[u] * wfL[(k + u) * 2 + fj];
      }
      for (int k = 0; k < HID; k += 8) {
        v8bf cv = *(const v8bf*)&hLDS[frow * LDH + k];
#pragma unroll
        for (int u = 0; u < 8; ++u) accf += (float)cv[u] * wfL[(IN_DIM + k + u) * 2 + fj];
      }
      aL[frow * 2 + fj] = (cosf(accf + bfj + phij) + 1.0f) * 0.5f;
    }
    __syncthreads();
    if (tid < BATCH) fL[tid] = 1.0f / (1.0f + __expf(aL[tid * 2 + 1] - aL[tid * 2 + 0]));
    __syncthreads();

    // ---- WMMA GEMM: (16x768) @ (768x96) per wave, bf16 in / f32 acc ----
    v8f acc[6];
#pragma unroll
    for (int n = 0; n < 6; ++n)
#pragma unroll
      for (int r = 0; r < 8; ++r) acc[n][r] = 0.f;

    const __bf16* xrow = xLDS + (size_t)(wave * 16 + nlo) * LDX;
    const __bf16* hrow = hLDS + (size_t)(wave * 16 + nlo) * LDH;
    for (int kt = 0; kt < KT; ++kt) {
      const __bf16* abase =
          (kt < 8) ? (xrow + kt * 32 + khalf) : (hrow + (kt - 8) * 32 + khalf);
      v8bf alo = *(const v8bf*)&abase[0];
      v8bf ahi = *(const v8bf*)&abase[8];
      v16bf a;
#pragma unroll
      for (int u = 0; u < 8; ++u) { a[u] = alo[u]; a[8 + u] = ahi[u]; }
#pragma unroll
      for (int n = 0; n < 6; ++n) {
        v16bf b = *(const v16bf*)&Wfrag[((size_t)(gnt[n] * KT + kt) * 512) + lane * 16];
        acc[n] = __builtin_amdgcn_wmma_f32_16x16x32_bf16(
            false, a, false, b, (short)0, acc[n], false, false);
      }
    }

    // ---- gates + cell update (all in C/D register layout) ----
    float fr[8];
#pragma unroll
    for (int r = 0; r < 8; ++r) fr[r] = fL[wave * 16 + r + mrow_off];

    __bf16* hnext = hb + (size_t)((t + 1) & 1) * BATCH * HID;
    float* outT = out + (size_t)t * BATCH * HID;
#pragma unroll
    for (int sub = 0; sub < 2; ++sub) {
      const int col = g * 32 + sub * 16 + nlo;
#pragma unroll
      for (int r = 0; r < 8; ++r) {
        float iv = acc[0 + sub][r] + biasReg[0 + sub];
        float uv = acc[2 + sub][r] + biasReg[2 + sub];
        float ov = acc[4 + sub][r] + biasReg[4 + sub];
        float ig = 1.f / (1.f + __expf(-iv));
        float gg = tanhf(uv);
        float og = 1.f / (1.f + __expf(-ov));
        float c = fr[r] * cst[sub][r] + ig * gg;
        cst[sub][r] = c;
        float h = og * tanhf(c);
        const int row = wave * 16 + r + mrow_off;
        outT[(size_t)row * HID + col] = h;                 // stacked output
        hnext[(size_t)row * HID + col] = (__bf16)h;        // next-step state
        if (t == T_STEPS - 1)                              // final hx
          out[(size_t)T_STEPS * BATCH * HID + (size_t)row * HID + col] = h;
      }
    }

    // ---- device-wide step boundary ----
    grid_sync(bar, bar + 16, NWG);
    // drop any stale lines from this WGP's read cache before next h read
    asm volatile("global_inv scope:SCOPE_DEV\n\ts_wait_loadcnt 0x0" ::: "memory");
  }

  // final cx
  const size_t cxBase = (size_t)T_STEPS * BATCH * HID + (size_t)BATCH * HID;
#pragma unroll
  for (int sub = 0; sub < 2; ++sub) {
    const int col = g * 32 + sub * 16 + nlo;
#pragma unroll
    for (int r = 0; r < 8; ++r) {
      const int row = wave * 16 + r + mrow_off;
      out[cxBase + (size_t)row * HID + col] = cst[sub][r];
    }
  }
}

// =====================================================================
extern "C" void kernel_launch(void* const* d_in, const int* in_sizes, int n_in,
                              void* d_out, int out_size, void* d_ws, size_t ws_size,
                              hipStream_t stream) {
  const float* X   = (const float*)d_in[0];
  const float* Wf  = (const float*)d_in[1];
  const float* bfv = (const float*)d_in[2];
  const float* Wi  = (const float*)d_in[3];
  const float* bi  = (const float*)d_in[4];
  const float* Wu  = (const float*)d_in[5];
  const float* bu  = (const float*)d_in[6];
  const float* Wo  = (const float*)d_in[7];
  const float* bo  = (const float*)d_in[8];
  const float* phi = (const float*)d_in[9];
  float* out = (float*)d_out;

  char* ws = (char*)d_ws;  // requires ~72 MB of workspace
  unsigned short* Wfrag = (unsigned short*)(ws + W_OFF);
  unsigned short* Xb    = (unsigned short*)(ws + X_OFF);
  unsigned short* hbuf  = (unsigned short*)(ws + H_OFF);
  unsigned int*   bar   = (unsigned int*)(ws + BAR_OFF);

  qlstm_prep<<<dim3(4096), dim3(256), 0, stream>>>(X, Wi, Wu, Wo, Wfrag, Xb, hbuf, bar);

  qlstm_rnn<<<dim3(NWG), dim3(256), LDS_TOTAL, stream>>>(
      Wfrag, Xb, hbuf, bar, Wf, bfv, bi, bu, bo, phi, out);
}
